// BrickVectorEdgeModel_34737695490544
// MI455X (gfx1250) — matrix operand order repaired and checked by reading, not compiled, not run on hardware
//
#include <hip/hip_runtime.h>

// ---------------------------------------------------------------------------
// BrickVectorEdgeModel, fused for MI455X (gfx1250, wave32):
//   WMMA f16->f32 16x16x32 GEMM core + Tensor Data Mover LDS staging.
// ---------------------------------------------------------------------------

#define HDIM   512
#define BATCH  4
#define NPTS   192
#define MNODE  (BATCH * NPTS)   // 768
#define KAUG   544              // 512 + 2 (xy) + 1 (bias) padded to mult of 32

typedef _Float16 v4h  __attribute__((ext_vector_type(4)));
typedef _Float16 v8h  __attribute__((ext_vector_type(8)));
typedef _Float16 v16h __attribute__((ext_vector_type(16)));
typedef float    v8f  __attribute__((ext_vector_type(8)));
typedef unsigned int u32x4 __attribute__((ext_vector_type(4)));
typedef int          i32x4 __attribute__((ext_vector_type(4)));
typedef int          i32x8 __attribute__((ext_vector_type(8)));

static __device__ inline v16h cat8(v8h lo, v8h hi) {
    return __builtin_shufflevector(lo, hi, 0,1,2,3,4,5,6,7,8,9,10,11,12,13,14,15);
}

// A fragment (16x32 f16, ISA 7.12.2): lanes<16 hold K = k0+[0..7] and k0+[16..23],
// lanes>=16 hold K = k0+[8..15] and k0+[24..31]; row M = lane%16.
static __device__ inline v16h load_a_frag(const _Float16* A, int sA, int m0, int k0, int lane) {
    int m  = m0 + (lane & 15);
    int kA = k0 + ((lane >> 4) << 3);
    const _Float16* p = A + (size_t)m * sA + kA;
    v8h lo = *(const v8h*)p;
    v8h hi = *(const v8h*)(p + 16);
    return cat8(lo, hi);
}

// B fragment (32x16 f16): W stored [N][K]; lane column n = n0 + lane%16,
// K = k0 + 8*(lane/16) + [0..7] and k0 + 16 + 8*(lane/16) + [0..7] (matches A).
static __device__ inline v16h load_b_frag(const _Float16* W, int sW, int n0, int k0, int lane) {
    int n  = n0 + (lane & 15);
    int kB = k0 + ((lane >> 4) << 3);
    const _Float16* p = W + (size_t)n * sW + kB;
    v8h lo = *(const v8h*)p;
    v8h hi = *(const v8h*)(p + 16);
    return cat8(lo, hi);
}

// ---------------------------------------------------------------------------
// Tensor Data Mover: stage one 512x32 f16 weight K-chunk (32 KB) into LDS.
// D# per CDNA5 ISA ch.8: group0 = count/lds_addr/global_addr/type=2,
// group1 = data_size=2B, tensor_dim0=tensor_dim1=512, tile_dim0=32,
// tile_dim1=512, tensor_dim0_stride=512. Groups 2/3 zero (2D tile).
// This toolchain exposes the 6-arg builtin (extra int32x8 group + cpol).
// ---------------------------------------------------------------------------

static __device__ inline void tdm_stage_chunk(const _Float16* Wg, int kb, _Float16* ldsDst, int tid) {
#if __has_builtin(__builtin_amdgcn_tensor_load_to_lds)
    unsigned long long ga = (unsigned long long)(uintptr_t)(const void*)(Wg + (size_t)kb * 32);
    unsigned int ldsOff   = (unsigned int)(uintptr_t)(void*)ldsDst;  // flat LDS addr[31:0]
    u32x4 g0;
    g0.x = 1u;                                       // count = 1 valid descriptor
    g0.y = ldsOff;                                   // lds_addr (bytes)
    g0.z = (unsigned int)ga;                         // global_addr[31:0]
    g0.w = (unsigned int)(ga >> 32) | (2u << 30);    // global_addr[56:32] | type=2
    i32x8 g1;
    g1[0] = 0x00010000;     // workgroup_mask=0, data_size=1 (2 bytes)
    g1[1] = 512 << 16;      // tensor_dim0 = 512 (bits 63:48)
    g1[2] = 512 << 16;      // tensor_dim0 hi=0 | tensor_dim1 = 512 (bits 95:80)
    g1[3] = 32 << 16;       // tensor_dim1 hi=0 | tile_dim0 = 32 (bits 127:112)
    g1[4] = 512;            // tile_dim1 = 512, tile_dim2 = 0
    g1[5] = 512;            // tensor_dim0_stride = 512 (bits 191:160)
    g1[6] = 0;
    g1[7] = 0;
    i32x4 gz4 = {0, 0, 0, 0};
    i32x8 gz8 = {0, 0, 0, 0, 0, 0, 0, 0};
    __builtin_amdgcn_tensor_load_to_lds(g0, g1, gz4, gz4, gz8, 0);
#else
    // Fallback: wave0 copies 512 rows x 32 B (first half of each 64 B row) cooperatively.
    int lane = tid & 31;
    for (int r = lane; r < HDIM; r += 32) {
        const uint4* src = (const uint4*)(Wg + (size_t)r * HDIM + kb * 32);
        uint4* dst = (uint4*)(ldsDst + r * 32);
        dst[0] = src[0];
        dst[1] = src[1];
        if (kb + 1 < 16)
            __builtin_prefetch(Wg + (size_t)r * HDIM + (kb + 1) * 32, 0, 1);
    }
#endif
}

// ---------------------------------------------------------------------------
// Prep kernels (tiny, run once per launch; f16 weights become L2-resident)
// ---------------------------------------------------------------------------

__global__ void build_a1(const float* __restrict__ brick, const float* __restrict__ xy,
                         _Float16* __restrict__ A1) {
    int i = blockIdx.x * 256 + threadIdx.x;           // over 768*544
    if (i >= MNODE * KAUG) return;
    int m = i / KAUG, k = i % KAUG;
    float val;
    if      (k < 512)  val = brick[(size_t)m * 512 + k];
    else if (k == 512) val = xy[m * 2 + 0];
    else if (k == 513) val = xy[m * 2 + 1];
    else if (k == 514) val = 1.0f;
    else               val = 0.0f;
    A1[i] = (_Float16)val;
}

__global__ void build_wcat(const float* __restrict__ W_a, const float* __restrict__ W_xy,
                           const float* __restrict__ b_a, const float* __restrict__ b_xy,
                           _Float16* __restrict__ Wcat) {
    int i = blockIdx.x * 256 + threadIdx.x;           // over 512*544
    if (i >= HDIM * KAUG) return;
    int g = i / KAUG, k = i % KAUG;
    float val;
    if      (k < 512)  val = W_a[(size_t)g * 512 + k];
    else if (k == 512) val = W_xy[g * 2 + 0];
    else if (k == 513) val = W_xy[g * 2 + 1];
    else if (k == 514) val = b_a[g] + b_xy[g];
    else               val = 0.0f;
    Wcat[i] = (_Float16)val;
}

__global__ void f32_to_f16(const float* __restrict__ s, _Float16* __restrict__ d, int n) {
    int i = blockIdx.x * 256 + threadIdx.x;
    if (i < n) d[i] = (_Float16)s[i];
}

__global__ void split_wca(const float* __restrict__ Wca,
                          _Float16* __restrict__ W1, _Float16* __restrict__ W2) {
    int i = blockIdx.x * 256 + threadIdx.x;           // over 512*512
    if (i >= HDIM * HDIM) return;
    int g = i >> 9, h = i & 511;
    W1[i] = (_Float16)Wca[(size_t)g * 1024 + h];
    W2[i] = (_Float16)Wca[(size_t)g * 1024 + 512 + h];
}

// ---------------------------------------------------------------------------
// Node GEMM: out = act(A[M x K] @ W[512 x K]^T + bias).
// 512 threads = 16 waves: 4 row-groups x 4 column quarters; 8 acc tiles/wave
// (64 VGPRs) leaves room to double-buffer B fragments for load/WMMA overlap.
// MODE 0: relu -> f16 out;  MODE 1: linear -> f32 out.
// ---------------------------------------------------------------------------

template <int MODE>
__global__ void __launch_bounds__(512)
node_gemm(const _Float16* __restrict__ A, int K, const _Float16* __restrict__ W,
          const float* __restrict__ bias, _Float16* __restrict__ outH,
          float* __restrict__ outF) {
    int tid  = threadIdx.x;
    int wave = tid >> 5, lane = tid & 31;
    int m0 = blockIdx.x * 64 + (wave >> 2) * 16;
    int n0 = (wave & 3) * 128;

    v8f acc[8];
#pragma unroll
    for (int nt = 0; nt < 8; ++nt) {
        float bv = bias ? bias[n0 + nt * 16 + (lane & 15)] : 0.0f;
        v8f t = {bv, bv, bv, bv, bv, bv, bv, bv};
        acc[nt] = t;
    }

    int KB = K >> 5;
    for (int kb = 0; kb < KB; ++kb) {
        v16h a = load_a_frag(A, K, m0, kb * 32, lane);
        v16h bcur = load_b_frag(W, K, n0, kb * 32, lane);
#pragma unroll
        for (int nt = 0; nt < 8; ++nt) {
            v16h bnext = bcur;
            if (nt < 7) bnext = load_b_frag(W, K, n0 + (nt + 1) * 16, kb * 32, lane);
            acc[nt] = __builtin_amdgcn_wmma_f32_16x16x32_f16(
                false, a, false, bcur, (short)0, acc[nt], false, false);
            bcur = bnext;
        }
    }

#pragma unroll
    for (int nt = 0; nt < 8; ++nt) {
        int n = n0 + nt * 16 + (lane & 15);
#pragma unroll
        for (int r = 0; r < 8; ++r) {
            int m = m0 + r + ((lane >> 4) << 3);
            float val = acc[nt][r];
            if (MODE == 0) outH[(size_t)m * HDIM + n] = (_Float16)fmaxf(val, 0.0f);
            else           outF[(size_t)m * HDIM + n] = val;
        }
    }
}

// ---------------------------------------------------------------------------
// Fused edge MLP. One WG = 64 edges (4 i x 16 j), 512 threads = 16 waves.
// LDS: 64x512 f16 activation buffer (reused across layers, 64 KB) +
// two 512x32 f16 weight chunks (2 x 32 KB) double-buffered via the TDM.
// ---------------------------------------------------------------------------

static __device__ inline void wg_layer(_Float16* Abuf, _Float16* WbufA, _Float16* WbufB,
                                       const _Float16* __restrict__ Wg,
                                       const float* __restrict__ bias, int tid) {
    int wave = tid >> 5, lane = tid & 31;
    int m0 = (wave >> 2) * 16;
    int n0 = (wave & 3) * 128;

    v8f acc[8];
#pragma unroll
    for (int nt = 0; nt < 8; ++nt) {
        float bv = bias[n0 + nt * 16 + (lane & 15)];
        v8f t = {bv, bv, bv, bv, bv, bv, bv, bv};
        acc[nt] = t;
    }

    // Prologue: chunk 0 in flight.
    if (wave == 0) tdm_stage_chunk(Wg, 0, WbufA, tid);

    for (int kb = 0; kb < 16; ++kb) {
        __syncthreads();   // compute kb-1 done everywhere -> buffer (kb+1)&1 is free
        _Float16* cur = (kb & 1) ? WbufB : WbufA;
        _Float16* nxt = (kb & 1) ? WbufA : WbufB;
        if (wave == 0) {
            if (kb < 15) {
                tdm_stage_chunk(Wg, kb + 1, nxt, tid);
                __builtin_amdgcn_s_wait_tensorcnt((short)1);  // chunk kb landed
            } else {
                __builtin_amdgcn_s_wait_tensorcnt((short)0);
            }
        }
        __syncthreads();   // chunk kb visible to all waves

        v16h a = load_a_frag(Abuf, HDIM, m0, kb * 32, lane);
        v16h bcur = load_b_frag(cur, 32, n0, 0, lane);
#pragma unroll
        for (int nt = 0; nt < 8; ++nt) {
            v16h bnext = bcur;
            if (nt < 7) bnext = load_b_frag(cur, 32, n0 + (nt + 1) * 16, 0, lane);
            acc[nt] = __builtin_amdgcn_wmma_f32_16x16x32_f16(
                false, a, false, bcur, (short)0, acc[nt], false, false);
            bcur = bnext;
        }
    }
    __syncthreads();       // all A reads done before overwrite
#pragma unroll
    for (int nt = 0; nt < 8; ++nt) {
        int n = n0 + nt * 16 + (lane & 15);
#pragma unroll
        for (int r = 0; r < 8; ++r) {
            int m = m0 + r + ((lane >> 4) << 3);
            Abuf[(size_t)m * HDIM + n] = (_Float16)fmaxf(acc[nt][r], 0.0f);
        }
    }
    __syncthreads();
}

__global__ void __launch_bounds__(512)
edge_kernel(const float* __restrict__ u, const float* __restrict__ v,
            const float* __restrict__ b_ca,
            const _Float16* __restrict__ Wcb, const float* __restrict__ b_cb,
            const _Float16* __restrict__ Wcc, const float* __restrict__ b_cc,
            const _Float16* __restrict__ Wout, const float* __restrict__ b_out,
            float* __restrict__ out) {
    extern __shared__ char smem[];
    _Float16* Abuf  = (_Float16*)smem;                          // 64 x 512 f16 (64 KB)
    _Float16* WbufA = (_Float16*)(smem + 64 * HDIM * 2);        // 512 x 32 f16 (32 KB)
    _Float16* WbufB = (_Float16*)(smem + 64 * HDIM * 2 + HDIM * 32 * 2);

    int tid = threadIdx.x;
    int b = blockIdx.z, ib = blockIdx.y, jb = blockIdx.x;
    int i0 = ib * 4, j0 = jb * 16;
    const float* ub = u + ((size_t)(b * NPTS + i0)) * HDIM;
    const float* vb = v + ((size_t)(b * NPTS + j0)) * HDIM;

    // e0[m][k] = relu(u[i][k] + v[j][k] + b_ca[k]) -> f16 in LDS (m = li*16+lj)
    for (int idx = tid; idx < 64 * 128; idx += 512) {
        int m = idx >> 7;
        int k = (idx & 127) << 2;
        int li = m >> 4, lj = m & 15;
        float4 uu = *(const float4*)(ub + (size_t)li * HDIM + k);
        float4 vv = *(const float4*)(vb + (size_t)lj * HDIM + k);
        float4 cc = *(const float4*)(b_ca + k);
        v4h hv;
        hv[0] = (_Float16)fmaxf(uu.x + vv.x + cc.x, 0.0f);
        hv[1] = (_Float16)fmaxf(uu.y + vv.y + cc.y, 0.0f);
        hv[2] = (_Float16)fmaxf(uu.z + vv.z + cc.z, 0.0f);
        hv[3] = (_Float16)fmaxf(uu.w + vv.w + cc.w, 0.0f);
        *(v4h*)(Abuf + (size_t)m * HDIM + k) = hv;
    }
    __syncthreads();

    wg_layer(Abuf, WbufA, WbufB, Wcb, b_cb, tid);   // e1 = relu(e0 @ Wcb^T + b_cb)
    wg_layer(Abuf, WbufA, WbufB, Wcc, b_cc, tid);   // e2 = relu(e1 @ Wcc^T + b_cc)

    // out[b, j, i, o] = e2 @ Wout^T + b_out   (128 scalar outputs per WG)
    if (tid < 128) {
        int m = tid >> 1, o = tid & 1;
        float s = b_out[o];
#pragma unroll 4
        for (int k8 = 0; k8 < 64; ++k8) {
            v8h aa = *(const v8h*)(Abuf + (size_t)m * HDIM + k8 * 8);
            v8h ww = *(const v8h*)(Wout + (size_t)o * HDIM + k8 * 8);
#pragma unroll
            for (int q = 0; q < 8; ++q) s += (float)aa[q] * (float)ww[q];
        }
        int li = m >> 4, lj = m & 15;
        out[(((size_t)b * NPTS + (j0 + lj)) * NPTS + (i0 + li)) * 2 + o] = s;
    }
}

// ---------------------------------------------------------------------------
// Host-side orchestration
// ---------------------------------------------------------------------------

extern "C" void kernel_launch(void* const* d_in, const int* in_sizes, int n_in,
                              void* d_out, int out_size, void* d_ws, size_t ws_size,
                              hipStream_t stream) {
    (void)in_sizes; (void)n_in; (void)out_size; (void)ws_size;

    const float* brick = (const float*)d_in[0];
    const float* xy    = (const float*)d_in[1];
    const float* W_xy  = (const float*)d_in[2];
    const float* b_xy  = (const float*)d_in[3];
    const float* W_a   = (const float*)d_in[4];
    const float* b_a   = (const float*)d_in[5];
    const float* W_b   = (const float*)d_in[6];
    const float* b_b   = (const float*)d_in[7];
    const float* W_ca  = (const float*)d_in[8];
    const float* b_ca  = (const float*)d_in[9];
    const float* W_cb  = (const float*)d_in[10];
    const float* b_cb  = (const float*)d_in[11];
    const float* W_cc  = (const float*)d_in[12];
    const float* b_cc  = (const float*)d_in[13];
    const float* W_out = (const float*)d_in[14];
    const float* b_out = (const float*)d_in[15];
    float* out = (float*)d_out;

    char* ws = (char*)d_ws;
    size_t off = 0;
    auto take = [&](size_t bytes) -> char* {
        off = (off + 255) & ~(size_t)255;
        char* p = ws + off;
        off += bytes;
        return p;
    };

    _Float16* A1    = (_Float16*)take((size_t)MNODE * KAUG * 2);
    _Float16* Wcat  = (_Float16*)take((size_t)HDIM * KAUG * 2);
    _Float16* f1    = (_Float16*)take((size_t)MNODE * HDIM * 2);
    _Float16* f2    = (_Float16*)take((size_t)MNODE * HDIM * 2);
    _Float16* Wb16  = (_Float16*)take((size_t)HDIM * HDIM * 2);
    _Float16* W1h   = (_Float16*)take((size_t)HDIM * HDIM * 2);
    _Float16* W2h   = (_Float16*)take((size_t)HDIM * HDIM * 2);
    _Float16* Wcb16 = (_Float16*)take((size_t)HDIM * HDIM * 2);
    _Float16* Wcc16 = (_Float16*)take((size_t)HDIM * HDIM * 2);
    _Float16* Wo16  = (_Float16*)take((size_t)2 * HDIM * 2);
    float*    ubuf  = (float*)take((size_t)MNODE * HDIM * 4);
    float*    vbuf  = (float*)take((size_t)MNODE * HDIM * 4);

    // --- prep: fold xy-path into augmented GEMM, f16-convert all weights ---
    build_a1<<<(MNODE * KAUG + 255) / 256, 256, 0, stream>>>(brick, xy, A1);
    build_wcat<<<(HDIM * KAUG + 255) / 256, 256, 0, stream>>>(W_a, W_xy, b_a, b_xy, Wcat);
    f32_to_f16<<<(HDIM * HDIM + 255) / 256, 256, 0, stream>>>(W_b, Wb16, HDIM * HDIM);
    f32_to_f16<<<(HDIM * HDIM + 255) / 256, 256, 0, stream>>>(W_cb, Wcb16, HDIM * HDIM);
    f32_to_f16<<<(HDIM * HDIM + 255) / 256, 256, 0, stream>>>(W_cc, Wcc16, HDIM * HDIM);
    f32_to_f16<<<(2 * HDIM + 255) / 256, 256, 0, stream>>>(W_out, Wo16, 2 * HDIM);
    split_wca<<<(HDIM * HDIM + 255) / 256, 256, 0, stream>>>(W_ca, W1h, W2h);

    // --- node stage: f1 -> f2 -> u, v (all WMMA) ---
    node_gemm<0><<<MNODE / 64, 512, 0, stream>>>(A1, KAUG, Wcat, (const float*)nullptr, f1, (float*)nullptr);
    node_gemm<0><<<MNODE / 64, 512, 0, stream>>>(f1, HDIM, Wb16, b_b, f2, (float*)nullptr);
    node_gemm<1><<<MNODE / 64, 512, 0, stream>>>(f2, HDIM, W1h, (const float*)nullptr, (_Float16*)nullptr, ubuf);
    node_gemm<1><<<MNODE / 64, 512, 0, stream>>>(f2, HDIM, W2h, (const float*)nullptr, (_Float16*)nullptr, vbuf);

    // --- fused edge stage: 2304 WGs, 128 KB dynamic LDS each ---
    size_t smem = (size_t)(64 * HDIM + 2 * HDIM * 32) * sizeof(_Float16);  // 131072 B
    dim3 egrid(NPTS / 16, NPTS / 4, BATCH);
    edge_kernel<<<egrid, 512, smem, stream>>>(ubuf, vbuf, b_ca, Wcb16, b_cb,
                                              Wcc16, b_cc, Wo16, b_out, out);
}
